// SparseEncoder_76192719831481
// MI455X (gfx1250) — matrix operand order repaired and checked by reading, not compiled, not run on hardware
//
#include <hip/hip_runtime.h>
#include <stdint.h>

// ---------------------------------------------------------------------------
// TopK Sparse Autoencoder forward for MI455X (gfx1250, wave32, WMMA bf16).
//   1) convert W_enc f32 -> bf16, activations f32 -> bf16      (ws)
//   2) transpose W_emb [d][m] -> bf16 [m][d]                   (ws)
//   3) fused encoder GEMM (v_wmma_f32_16x16x32_bf16) + streaming top-64;
//      A tile staged into LDS via GLOBAL_LOAD_ASYNC_TO_LDS_B128 (ASYNCcnt)
//   4) decoder: gather 64 bf16 rows of W_embT per token, fma into f32 out
//   5) finalize aux loss scalar
// ---------------------------------------------------------------------------

typedef __attribute__((ext_vector_type(16))) __bf16 v16bf;
typedef __attribute__((ext_vector_type(8)))  float  v8f;

#define M_TILE    32          // tokens per workgroup in encoder
#define KSEL      64          // top-k (reference topk = 64)
#define CAP       128         // candidate buffer per token (= cols/WG/iter)
#define AUX_COEFF 0.03125f

// ---- CDNA5 async global->LDS copy (16B per lane), tracked by ASYNCcnt ------
__device__ __forceinline__ void async_copy_b128(const void* g, void* l) {
  // LDS offset: flat -> AS(3) addrspacecast strips the aperture base.
  unsigned loff = (unsigned)(size_t)(__attribute__((address_space(3))) char*)l;
  unsigned long long ga = (unsigned long long)g;
  asm volatile("global_load_async_to_lds_b128 %0, %1, off"
               :: "v"(loff), "v"(ga) : "memory");
}
__device__ __forceinline__ void wait_asynccnt0() {
  asm volatile("s_wait_asynccnt 0x0" ::: "memory");
}

// ---------------- f32 -> bf16 elementwise convert ---------------------------
__global__ __launch_bounds__(256)
void k_convert_bf16(const float* __restrict__ src, uint16_t* __restrict__ dst,
                    long long n4) {
  long long i = (long long)blockIdx.x * blockDim.x + threadIdx.x;
  if (i >= n4) return;
  const float4 f = reinterpret_cast<const float4*>(src)[i];
  union { __bf16 h[4]; uint2 u; } cv;
  cv.h[0] = (__bf16)f.x; cv.h[1] = (__bf16)f.y;
  cv.h[2] = (__bf16)f.z; cv.h[3] = (__bf16)f.w;
  reinterpret_cast<uint2*>(dst)[i] = cv.u;
}

// ---------------- W_emb [d][m] f32  ->  W_embT [m][d] bf16 ------------------
__global__ __launch_bounds__(256)
void k_transpose_wemb(const float* __restrict__ W, uint16_t* __restrict__ WT,
                      int d, int m) {
  __shared__ float tile[32][33];
  const int mBase = blockIdx.x * 32;
  const int dBase = blockIdx.y * 32;
  const int tx = threadIdx.x & 31;
  const int ty = threadIdx.x >> 5;          // 0..7
#pragma unroll
  for (int i = 0; i < 4; ++i)
    tile[ty + 8 * i][tx] = W[(size_t)(dBase + ty + 8 * i) * m + mBase + tx];
  __syncthreads();
#pragma unroll
  for (int i = 0; i < 4; ++i) {
    union { __bf16 h; uint16_t u; } cv;
    cv.h = (__bf16)tile[tx][ty + 8 * i];
    WT[(size_t)(mBase + ty + 8 * i) * d + dBase + tx] = cv.u;
  }
}

// ---------------- tiny zero kernel (aux accumulator lives in ws) ------------
__global__ void k_zero(float* __restrict__ p, int n) {
  int i = blockIdx.x * blockDim.x + threadIdx.x;
  if (i < n) p[i] = 0.0f;
}

// ---------------- fused encoder GEMM + streaming top-k ----------------------
__global__ __launch_bounds__(256)
void k_encoder_topk(const uint16_t* __restrict__ actB,   // [BT][d] bf16
                    const uint16_t* __restrict__ WencB,  // [m][d] bf16
                    const float*    __restrict__ benc,   // [m]
                    const int*      __restrict__ steps,  // [m]
                    const int*      __restrict__ dwp,    // dead_window
                    float* __restrict__ tkv,             // [BT][KSEL]
                    int*   __restrict__ tki,             // [BT][KSEL]
                    float* __restrict__ aux_accum,       // [1]
                    int d, int m) {
  extern __shared__ char smem[];
  __bf16* A = (__bf16*)smem;                                // M_TILE * d bf16
  size_t o = (size_t)M_TILE * d * sizeof(__bf16);
  float* kept_v = (float*)(smem + o); o += (size_t)M_TILE * KSEL * 4;
  int*   kept_i = (int*)  (smem + o); o += (size_t)M_TILE * KSEL * 4;
  float* cand_v = (float*)(smem + o); o += (size_t)M_TILE * CAP * 4;
  int*   cand_i = (int*)  (smem + o); o += (size_t)M_TILE * CAP * 4;
  float* thr    = (float*)(smem + o); o += (size_t)M_TILE * 4;
  int*   cnt    = (int*)  (smem + o); o += (size_t)M_TILE * 4;
  float* red    = (float*)(smem + o);                       // 256 f32

  const int tid  = threadIdx.x;
  const int lane = tid & 31;
  const int wave = tid >> 5;          // 0..7
  const int tok0 = blockIdx.x * M_TILE;
  const int dw   = dwp[0];

  // Stage A tile into LDS via async copies (no VGPR round-trip).
  {
    const size_t tileBytes = (size_t)M_TILE * d * sizeof(__bf16);
    const char* gsrc = (const char*)(actB + (size_t)tok0 * d);
    for (size_t b = (size_t)tid * 16; b < tileBytes; b += 256 * 16)
      async_copy_b128(gsrc + b, (char*)smem + b);
  }
  if (tid < M_TILE) { thr[tid] = -3.0e38f; cnt[tid] = 0; }
  for (int i = tid; i < M_TILE * KSEL; i += 256) {
    kept_v[i] = -3.0e38f; kept_i[i] = 0;
  }
  wait_asynccnt0();
  __syncthreads();

  const __bf16* Wb = reinterpret_cast<const __bf16*>(WencB);
  const int nB = lane & 15;           // N within 16-wide tile / A row (M)
  const int g  = lane >> 4;           // K-half selector (ISA bf16 layouts)
  float deadsum = 0.0f;

  // Each wave sweeps 16 concept-columns per iteration; 8 waves -> 128 cols/WG.
  for (int n0 = wave * 16; n0 < m; n0 += 128) {
    const int n = n0 + nB;
    const __bf16* Brow = Wb + (size_t)n * d;
    if (n0 + 128 < m)
      __builtin_prefetch(Wb + (size_t)(n + 128) * d, 0, 1);  // global_prefetch_b8

    v8f acc0 = {};     // tokens 0..15 of tile
    v8f acc1 = {};     // tokens 16..31 of tile
    for (int k0 = 0; k0 < d; k0 += 32) {
      union { v16bf v; uint4 q[2]; } bfr, a0, a1;
      // B 32x16 bf16: lane (N=nB, g) holds K = k0+g*8+[0..7] and k0+16+g*8+[0..7]
      const __bf16* bp = Brow + k0 + g * 8;
      bfr.q[0] = *reinterpret_cast<const uint4*>(bp);
      bfr.q[1] = *reinterpret_cast<const uint4*>(bp + 16);
      // A 16x32 bf16: lane (M=nB, g) holds K = k0+g*8+[0..7] and k0+16+g*8+[0..7]
      const __bf16* ar0 = A + (size_t)nB * d + k0 + g * 8;
      a0.q[0] = *reinterpret_cast<const uint4*>(ar0);
      a0.q[1] = *reinterpret_cast<const uint4*>(ar0 + 16);
      const __bf16* ar1 = ar0 + (size_t)16 * d;
      a1.q[0] = *reinterpret_cast<const uint4*>(ar1);
      a1.q[1] = *reinterpret_cast<const uint4*>(ar1 + 16);

      acc0 = __builtin_amdgcn_wmma_f32_16x16x32_bf16(
          false, a0.v, false, bfr.v, (short)0, acc0, false, false);
      acc1 = __builtin_amdgcn_wmma_f32_16x16x32_bf16(
          false, a1.v, false, bfr.v, (short)0, acc1, false, false);
    }

    // pre_act values: C/D layout -> token = r + 8*g (+16 for acc1), col = n.
    const float bias = benc[n];
    const bool  dead = (steps[n] >= dw);
#pragma unroll
    for (int r = 0; r < 8; ++r) {
      const float v0 = acc0[r] + bias;
      const float v1 = acc1[r] + bias;
      const int t0 = r + 8 * g;
      const int t1 = t0 + 16;
      if (dead) deadsum += v0 + v1;
      if (v0 > thr[t0]) {
        int pos = atomicAdd(&cnt[t0], 1);
        if (pos < CAP) { cand_v[t0 * CAP + pos] = v0; cand_i[t0 * CAP + pos] = n; }
      }
      if (v1 > thr[t1]) {
        int pos = atomicAdd(&cnt[t1], 1);
        if (pos < CAP) { cand_v[t1 * CAP + pos] = v1; cand_i[t1 * CAP + pos] = n; }
      }
    }

    // Compact candidates into the per-token top-64 list, refresh threshold.
    __syncthreads();
    if (tid < M_TILE) {
      const int t = tid;
      int c = cnt[t]; if (c > CAP) c = CAP;
      float* kv = kept_v + t * KSEL;
      int*   ki = kept_i + t * KSEL;
      for (int i = 0; i < c; ++i) {
        const float v = cand_v[t * CAP + i];
        int mi = 0; float mv = kv[0];
        for (int j = 1; j < KSEL; ++j) {
          const float x = kv[j];
          if (x < mv) { mv = x; mi = j; }
        }
        if (v > mv) { kv[mi] = v; ki[mi] = cand_i[t * CAP + i]; }
      }
      float mv = kv[0];
      for (int j = 1; j < KSEL; ++j) mv = fminf(mv, kv[j]);
      thr[t] = mv;
      cnt[t] = 0;
    }
    __syncthreads();
  }

  // Emit top-k lists.
  for (int i = tid; i < M_TILE * KSEL; i += 256) {
    const int t = i >> 6, j = i & (KSEL - 1);
    tkv[(size_t)(tok0 + t) * KSEL + j] = kept_v[i];
    tki[(size_t)(tok0 + t) * KSEL + j] = kept_i[i];
  }
  // Dead-concept partial sum -> global accumulator.
  red[tid] = deadsum;
  __syncthreads();
  for (int s = 128; s > 0; s >>= 1) {
    if (tid < s) red[tid] += red[tid + s];
    __syncthreads();
  }
  if (tid == 0) atomicAdd(aux_accum, red[0]);
}

// ---------------- decoder: encoded[token] = sum_j val_j * W_embT[idx_j] -----
__global__ __launch_bounds__(256)
void k_decoder(const float* __restrict__ tkv, const int* __restrict__ tki,
               const uint16_t* __restrict__ WTb, float* __restrict__ out, int d) {
  __shared__ float sv[KSEL];
  __shared__ int   si[KSEL];
  const int token = blockIdx.x;
  const int tid   = threadIdx.x;
  if (tid < KSEL) {
    sv[tid] = tkv[(size_t)token * KSEL + tid];
    si[tid] = tki[(size_t)token * KSEL + tid];
  }
  __syncthreads();
  const __bf16* WT = reinterpret_cast<const __bf16*>(WTb);
  for (int col0 = tid * 8; col0 < d; col0 += 256 * 8) {
    float acc[8] = {0.f, 0.f, 0.f, 0.f, 0.f, 0.f, 0.f, 0.f};
    for (int j = 0; j < KSEL; ++j) {
      const float v  = sv[j];
      const int   ix = si[j];
      union { uint4 q; __bf16 h[8]; } w;
      w.q = *reinterpret_cast<const uint4*>(WT + (size_t)ix * d + col0);
#pragma unroll
      for (int e = 0; e < 8; ++e) acc[e] += v * (float)w.h[e];
    }
    float* op = out + (size_t)token * d + col0;
    float4 o0 = {acc[0], acc[1], acc[2], acc[3]};
    float4 o1 = {acc[4], acc[5], acc[6], acc[7]};
    *reinterpret_cast<float4*>(op)     = o0;
    *reinterpret_cast<float4*>(op + 4) = o1;
  }
}

// ---------------- aux loss scalar -------------------------------------------
__global__ __launch_bounds__(256)
void k_finalize(const int* __restrict__ steps, const int* __restrict__ dwp,
                const float* __restrict__ aux_accum, float* __restrict__ out_aux,
                int m, int BT) {
  __shared__ int sred[256];
  const int tid = threadIdx.x;
  const int dw  = dwp[0];
  int c = 0;
  for (int i = tid; i < m; i += 256) c += (steps[i] >= dw) ? 1 : 0;
  sred[tid] = c;
  __syncthreads();
  for (int s = 128; s > 0; s >>= 1) {
    if (tid < s) sred[tid] += sred[tid + s];
    __syncthreads();
  }
  if (tid == 0) {
    const int nd = sred[0];
    const float denom = fmaxf((float)nd * (float)BT, 1.0f);
    const float mean  = aux_accum[0] / denom;
    out_aux[0] = (nd > 0) ? (-mean * AUX_COEFF) : 0.0f;
  }
}

// ---------------------------------------------------------------------------
extern "C" void kernel_launch(void* const* d_in, const int* in_sizes, int n_in,
                              void* d_out, int out_size, void* d_ws, size_t ws_size,
                              hipStream_t stream) {
  const float* act  = (const float*)d_in[0];
  const float* Wenc = (const float*)d_in[1];
  const float* benc = (const float*)d_in[2];
  const float* Wemb = (const float*)d_in[3];
  const int*   steps= (const int*)  d_in[4];
  const int*   dwp  = (const int*)  d_in[6];   // dead_window (device scalar)

  const int m  = in_sizes[2];          // 16384
  const int d  = in_sizes[1] / m;      // 2048
  const int BT = in_sizes[0] / d;      // 4096

  char* ws = (char*)d_ws;
  size_t off = 0;
  uint16_t* WencB = (uint16_t*)(ws + off); off += (size_t)m * d * 2;    // 64 MB
  uint16_t* WTb   = (uint16_t*)(ws + off); off += (size_t)m * d * 2;    // 64 MB
  uint16_t* ActB  = (uint16_t*)(ws + off); off += (size_t)BT * d * 2;   // 16 MB
  float*    tkv   = (float*)   (ws + off); off += (size_t)BT * KSEL * 4;
  int*      tki   = (int*)     (ws + off); off += (size_t)BT * KSEL * 4;
  float*    aux   = (float*)   (ws + off); off += 16;

  float* enc_out = (float*)d_out;
  float* aux_out = enc_out + (size_t)BT * d;

  k_zero<<<1, 64, 0, stream>>>(aux, 1);

  {
    long long n4 = (long long)m * d / 4;
    k_convert_bf16<<<(int)((n4 + 255) / 256), 256, 0, stream>>>(Wenc, WencB, n4);
  }
  {
    long long n4 = (long long)BT * d / 4;
    k_convert_bf16<<<(int)((n4 + 255) / 256), 256, 0, stream>>>(act, ActB, n4);
  }
  {
    dim3 g(m / 32, d / 32);
    k_transpose_wemb<<<g, 256, 0, stream>>>(Wemb, WTb, d, m);
  }
  {
    size_t sh = (size_t)M_TILE * d * 2          // A tile (bf16)
              + (size_t)M_TILE * KSEL * 8       // kept val+idx
              + (size_t)M_TILE * CAP  * 8       // candidate val+idx
              + (size_t)M_TILE * 8              // thr + cnt
              + 256 * 4;                        // reduction
    k_encoder_topk<<<BT / M_TILE, 256, sh, stream>>>(
        ActB, WencB, benc, steps, dwp, tkv, tki, aux, d, m);
  }
  k_decoder<<<BT, 256, 0, stream>>>(tkv, tki, WTb, enc_out, d);
  k_finalize<<<1, 256, 0, stream>>>(steps, dwp, aux, aux_out, m, BT);
}
